// ConvDownsample2D_40604620817122
// MI455X (gfx1250) — matrix operand order repaired and checked by reading, not compiled, not run on hardware
//
#include <hip/hip_runtime.h>
#include <hip/hip_bf16.h>

typedef __attribute__((ext_vector_type(16))) _Float16 v16h;
typedef __attribute__((ext_vector_type(8)))  _Float16 v8h;
typedef __attribute__((ext_vector_type(4)))  _Float16 v4h;
typedef __attribute__((ext_vector_type(8)))  float    v8f;
typedef __attribute__((ext_vector_type(4)))  float    v4f;
typedef __attribute__((ext_vector_type(4)))  int      v4i;

typedef __attribute__((address_space(1))) v4i* gv4i_p;   // global int4*
typedef __attribute__((address_space(3))) v4i* lv4i_p;   // LDS int4*

#define BB   8
#define HH   256
#define WW   256
#define CI   128
#define CO   256
#define YH   257
#define YW   257
#define KTOT 1152                 // 3*3*128
#define NCHUNK 36                 // KTOT / 32
#define BROW 40                   // LDS B row stride in f16 (80 B, bank-conflict-free)

#if __has_builtin(__builtin_amdgcn_global_load_async_to_lds_b128)
#define HAVE_ASYNC_LDS 1
#else
#define HAVE_ASYNC_LDS 0
#endif

// ---------------------------------------------------------------------------
// Kernel 1: conv_w [3,3,128,256] (HWIO, fp32) -> wT[co][k] f16, k=(kh*3+kw)*128+ci
// ---------------------------------------------------------------------------
__global__ __launch_bounds__(256) void wcvt_kernel(const float* __restrict__ w,
                                                   _Float16* __restrict__ wT) {
    int idx = blockIdx.x * 256 + threadIdx.x;          // idx = k*256 + co
    if (idx >= 9 * CI * CO) return;
    int co = idx & (CO - 1);
    int k  = idx >> 8;
    wT[(size_t)co * KTOT + k] = (_Float16)w[idx];
}

// ---------------------------------------------------------------------------
// Kernel 2: depthwise 4x4 FIR blur, pad (2,2), fp32 -> f16 (NHWC)
// ---------------------------------------------------------------------------
__global__ __launch_bounds__(256) void blur_kernel(const float* __restrict__ x,
                                                   const float* __restrict__ fir,
                                                   _Float16* __restrict__ y) {
    int idx = blockIdx.x * 256 + threadIdx.x;
    const int total = BB * YH * YW * (CI / 4);
    if (idx >= total) return;

    int c4 = idx & 31;
    int t  = idx >> 5;
    int w  = t % YW;  t /= YW;
    int h  = t % YH;
    int b  = t / YH;

    float fk[4] = { fir[0], fir[1], fir[2], fir[3] };
    float s   = fk[0] + fk[1] + fk[2] + fk[3];
    float inv = 1.0f / (s * s);                        // GAIN = 1.0

    v4f acc = { 0.f, 0.f, 0.f, 0.f };
    #pragma unroll
    for (int a = 0; a < 4; ++a) {
        int ih = h + a - 2;
        if (ih < 0 || ih >= HH) continue;
        #pragma unroll
        for (int d = 0; d < 4; ++d) {
            int iw = w + d - 2;
            if (iw < 0 || iw >= WW) continue;
            const v4f v = *(const v4f*)(x + (((size_t)b * HH + ih) * WW + iw) * CI + c4 * 4);
            acc += (fk[a] * fk[d] * inv) * v;
        }
    }
    v4h o;
    #pragma unroll
    for (int i = 0; i < 4; ++i) o[i] = (_Float16)acc[i];
    *(v4h*)(y + (((size_t)b * YH + h) * YW + w) * CI + c4 * 4) = o;
}

// ---------------------------------------------------------------------------
// Kernel 3: implicit-GEMM conv via V_WMMA_F32_16X16X32_F16.
// M = 8*128*128 (b,oh,ow), N = 256 (co), K = 1152.
// Block = 256 threads = 8 waves. Tile: M=128 (16 rows/wave), N=64 (4 accums).
// B tile (64x32 f16) staged through double-buffered LDS with async copies;
// A prefetched into registers one chunk ahead.
// ---------------------------------------------------------------------------
__global__ __launch_bounds__(256) void gemm_kernel(const _Float16* __restrict__ y,
                                                   const _Float16* __restrict__ wT,
                                                   const float* __restrict__ bias,
                                                   float* __restrict__ out) {
    __shared__ _Float16 bTile[2][64][BROW];            // 2 x 5120 B

    const int lane  = threadIdx.x & 31;
    const int wave  = threadIdx.x >> 5;
    const int l15   = lane & 15;
    const int hi    = lane >> 4;
    const int mBase = blockIdx.x * 128 + wave * 16;
    const int nBase = blockIdx.y * 64;

    // ---- A row (M) decomposition: m = ((b*128)+oh)*128 + ow
    const int m  = mBase + l15;
    const int b  = m >> 14;
    const int oh = (m >> 7) & 127;
    const int ow = m & 127;
    const size_t aRow0 = (((size_t)b * YH + 2 * oh) * YW + 2 * ow) * CI;

    const int kHalfA = hi ? 8 : 0;    // A layout: lanes16-31 take K+8 / K+24
    const int kHalfB = hi ? 8 : 0;    // B layout: lanes16-31 take K+8 / K+24

    // ---- B staging assignment: thread -> (row n, 16B segment)
    const int sN   = threadIdx.x >> 2;                 // 0..63
    const int sSeg = threadIdx.x & 3;                  // 0..3
    const _Float16* gB = wT + (size_t)(nBase + sN) * KTOT + sSeg * 8;

    float biasv[4];
    #pragma unroll
    for (int t = 0; t < 4; ++t) biasv[t] = bias[nBase + t * 16 + l15];

    auto stage = [&](int p, int buf) {                 // copy chunk p's B tile to LDS
        const _Float16* src = gB + p * 32;
        _Float16* dst = &bTile[buf][sN][sSeg * 8];
#if HAVE_ASYNC_LDS
        __builtin_amdgcn_global_load_async_to_lds_b128(
            (gv4i_p)(void*)src, (lv4i_p)(void*)dst, 0, 0);
#else
        v8h r = *(const v8h*)src;
        *(v8h*)dst = r;
#endif
    };
    auto wait_stage = [&]() {
#if HAVE_ASYNC_LDS
#if __has_builtin(__builtin_amdgcn_s_wait_asynccnt)
        __builtin_amdgcn_s_wait_asynccnt(0);
#else
        asm volatile("s_wait_asynccnt 0" ::: "memory");
#endif
#endif
    };

    // ---- A address for chunk p (per-lane, two 16B segments at +0 / +16)
    auto aAddr = [&](int p) -> const _Float16* {
        const int kc   = p * 32;
        const int khkw = kc >> 7;                      // 0..8; const per chunk
        const int kh = khkw / 3, kw = khkw % 3;
        return y + aRow0 + (size_t)(kh * YW + kw) * CI + (kc & 127) + kHalfA;
    };

    // ---- prologue: stage chunk 0, prefetch A chunk 0
    stage(0, 0);
    v8h a0n = *(const v8h*)aAddr(0);
    v8h a1n = *(const v8h*)(aAddr(0) + 16);
    wait_stage();
    __syncthreads();

    v8f acc[4] = {};

    for (int p = 0; p < NCHUNK; ++p) {
        const int buf = p & 1;

        // 1) kick off async stage of next B tile into the other buffer
        if (p + 1 < NCHUNK) stage(p + 1, buf ^ 1);

        // 2) rotate A regs; prefetch A for next chunk
        v8h a0c = a0n, a1c = a1n;
        if (p + 1 < NCHUNK) {
            const _Float16* ap = aAddr(p + 1);
            a0n = *(const v8h*)ap;
            a1n = *(const v8h*)(ap + 16);
        }
        v16h av = __builtin_shufflevector(a0c, a1c,
                    0,1,2,3,4,5,6,7,8,9,10,11,12,13,14,15);

        // 3) B from LDS (current buffer) + 4 WMMAs
        #pragma unroll
        for (int t = 0; t < 4; ++t) {
            const _Float16* bp = &bTile[buf][t * 16 + l15][kHalfB];
            v8h b0 = *(const v8h*)(bp);
            v8h b1 = *(const v8h*)(bp + 16);
            v16h bv = __builtin_shufflevector(b0, b1,
                        0,1,2,3,4,5,6,7,8,9,10,11,12,13,14,15);
            acc[t] = __builtin_amdgcn_wmma_f32_16x16x32_f16(
                false, av, false, bv, (short)0, acc[t], false, false);
        }

        // 4) next tile staged + everyone done reading current buffer
        wait_stage();
        __syncthreads();
    }

    // ---- epilogue: C/D layout — VGPR j: M = mBase + j + 8*hi, N = nBase+t*16+l15
    #pragma unroll
    for (int t = 0; t < 4; ++t) {
        const int n = nBase + t * 16 + l15;
        #pragma unroll
        for (int j = 0; j < 8; ++j) {
            const int mRow = mBase + j + hi * 8;
            out[(size_t)mRow * CO + n] = acc[t][j] + biasv[t];
        }
    }
}

// ---------------------------------------------------------------------------
extern "C" void kernel_launch(void* const* d_in, const int* in_sizes, int n_in,
                              void* d_out, int out_size, void* d_ws, size_t ws_size,
                              hipStream_t stream) {
    const float* x      = (const float*)d_in[0];   // [8,256,256,128]
    const float* conv_w = (const float*)d_in[1];   // [3,3,128,256]
    const float* conv_b = (const float*)d_in[2];   // [256]
    const float* fir    = (const float*)d_in[3];   // [4]
    float* out = (float*)d_out;                    // [8,128,128,256]

    _Float16* wT = (_Float16*)d_ws;                              // 0.59 MB
    _Float16* y  = (_Float16*)((char*)d_ws + (size_t)(4 << 20)); // 135.3 MB @ +4MB

    {   // weight convert + transpose
        int n = 9 * CI * CO;
        wcvt_kernel<<<(n + 255) / 256, 256, 0, stream>>>(conv_w, wT);
    }
    {   // depthwise FIR blur
        int n = BB * YH * YW * (CI / 4);
        blur_kernel<<<(n + 255) / 256, 256, 0, stream>>>(x, fir, y);
    }
    {   // WMMA implicit GEMM + bias
        dim3 grid(1024, 4);                        // M/128 = 1024, N/64 = 4
        gemm_kernel<<<grid, 256, 0, stream>>>(y, wT, conv_b, out);
    }
}
